// QuantMatMul_32856499815108
// MI455X (gfx1250) — 8-bit matrix operand order repaired and checked by reading, compiled, not run on hardware
//
#include <hip/hip_runtime.h>

typedef __attribute__((ext_vector_type(16))) int   v16i;
typedef __attribute__((ext_vector_type(8)))  float v8f;
typedef __attribute__((ext_vector_type(4)))  int   v4i;

#if __has_builtin(__builtin_amdgcn_global_load_async_to_lds_b128)
#define QMM_HAVE_ASYNC_LDS 1
typedef __attribute__((address_space(1))) v4i GV4;
typedef __attribute__((address_space(3))) v4i LV4;
#endif

__device__ __forceinline__ void qmm_wait_async_8() {
#if defined(QMM_HAVE_ASYNC_LDS)
#if __has_builtin(__builtin_amdgcn_s_wait_asynccnt)
    __builtin_amdgcn_s_wait_asynccnt(8);
#else
    asm volatile("s_wait_asynccnt 0x8" ::: "memory");
#endif
#endif
}
__device__ __forceinline__ void qmm_wait_async_0() {
#if defined(QMM_HAVE_ASYNC_LDS)
#if __has_builtin(__builtin_amdgcn_s_wait_asynccnt)
    __builtin_amdgcn_s_wait_asynccnt(0);
#else
    asm volatile("s_wait_asynccnt 0x0" ::: "memory");
#endif
#endif
}

__device__ __forceinline__ unsigned char f32_to_e4m3(float f) {
    unsigned int fb   = __float_as_uint(f);
    unsigned int sign = fb & 0x80000000u;
    fb ^= sign;
    unsigned char result;
    if (fb >= (1087u << 20)) {
        result = 0x7f;
    } else if (fb < (121u << 23)) {
        const unsigned int denorm_mask = 141u << 23;
        float z = __uint_as_float(fb) + __uint_as_float(denorm_mask);
        result  = (unsigned char)(__float_as_uint(z) - denorm_mask);
    } else {
        unsigned int mant_odd = (fb >> 20) & 1u;
        fb += ((unsigned int)(7 - 127) << 23) + 0x7FFFFu;
        fb += mant_odd;
        result = (unsigned char)(fb >> 20);
    }
    return result | (unsigned char)(sign >> 24);
}

__device__ __forceinline__ float slot_scale(const unsigned* amax_slot) {
    return fmaxf(__uint_as_float(*amax_slot) / 448.0f, 1e-12f);
}

__global__ void qmm_init(unsigned* ws) {
    if (threadIdx.x < 2) ws[threadIdx.x] = 0u;
}

__global__ void qmm_amax(const float4* __restrict__ x, long n4, unsigned* amax_slot) {
    float m = 0.0f;
    long stride = (long)gridDim.x * blockDim.x;
    for (long i = (long)blockIdx.x * blockDim.x + threadIdx.x; i < n4; i += stride) {
        float4 v = x[i];
        m = fmaxf(m, fmaxf(fmaxf(fabsf(v.x), fabsf(v.y)),
                           fmaxf(fabsf(v.z), fabsf(v.w))));
    }
    #pragma unroll
    for (int off = 16; off >= 1; off >>= 1)
        m = fmaxf(m, __shfl_xor(m, off, 32));
    if ((threadIdx.x & 31) == 0)
        atomicMax(amax_slot, __float_as_uint(m));
}

__global__ void qmm_quant(const float4* __restrict__ x, unsigned char* __restrict__ q,
                          const unsigned* __restrict__ amax_slot, long n16) {
    long i = (long)blockIdx.x * blockDim.x + threadIdx.x;
    if (i >= n16) return;
    float inv = 1.0f / slot_scale(amax_slot);
    unsigned w[4];
    #pragma unroll
    for (int j = 0; j < 4; ++j) {
        float4 v = x[i * 4 + j];
        w[j] =  (unsigned)f32_to_e4m3(v.x * inv)
             | ((unsigned)f32_to_e4m3(v.y * inv) << 8)
             | ((unsigned)f32_to_e4m3(v.z * inv) << 16)
             | ((unsigned)f32_to_e4m3(v.w * inv) << 24);
    }
    ((uint4*)q)[i] = make_uint4(w[0], w[1], w[2], w[3]);
}

__global__ void qmm_quant_t(const float* __restrict__ x, unsigned char* __restrict__ qt,
                            const unsigned* __restrict__ amax_slot, int K, int N) {
    __shared__ unsigned char tile[32][33];
    float inv = 1.0f / slot_scale(amax_slot);
    int tileK = blockIdx.y * 32;
    int tileN = blockIdx.x * 32;
    int tx = threadIdx.x, ty = threadIdx.y;
    #pragma unroll
    for (int i = 0; i < 4; ++i) {
        int k = tileK + ty + i * 8;
        int n = tileN + tx;
        tile[ty + i * 8][tx] = f32_to_e4m3(x[(size_t)k * N + n] * inv);
    }
    __syncthreads();
    #pragma unroll
    for (int i = 0; i < 4; ++i) {
        int n = tileN + ty + i * 8;
        int k = tileK + tx;
        qt[(size_t)n * K + k] = tile[tx][ty + i * 8];
    }
}

__global__ __launch_bounds__(256) void qmm_gemm(
    const unsigned char* __restrict__ qA,
    const unsigned char* __restrict__ qBT,
    float* __restrict__ out,
    const unsigned* __restrict__ amax, int M, int N, int K) {

    __shared__ __align__(16) unsigned char ldsA[2][128 * 128];
    __shared__ __align__(16) unsigned char ldsB[2][128 * 128];

    const int t     = threadIdx.x;
    const int lane  = t & 31;
    const int wave  = t >> 5;
    const int row16 = lane & 15;
    const int hl    = lane >> 4;

    const int nBase = blockIdx.x * 128;
    const int mBase = blockIdx.y * 128;
    const size_t batchA = (size_t)blockIdx.z * M * K;
    const size_t batchC = (size_t)blockIdx.z * M * N;

    const int wm = (wave >> 2) * 4;
    const int wn = (wave & 3) * 2;

    const unsigned char* gA[4];
    const unsigned char* gB[4];
    int soff[4];
    #pragma unroll
    for (int i = 0; i < 4; ++i) {
        int idx = t + i * 256;
        int r   = idx >> 3;
        int c   = (idx & 7) * 16;
        soff[i] = r * 128 + c;
        gA[i]   = qA + batchA + (size_t)(mBase + r) * K + c;
        gB[i]   = qBT + (size_t)(nBase + r) * K + c;
    }

    v8f acc[4][2];
    #pragma unroll
    for (int i = 0; i < 4; ++i)
        #pragma unroll
        for (int j = 0; j < 2; ++j)
            acc[i][j] = (v8f){0.f, 0.f, 0.f, 0.f, 0.f, 0.f, 0.f, 0.f};

    const int nChunks = K >> 7;

    auto stage = [&](int kb, int buf) {
        #pragma unroll
        for (int i = 0; i < 4; ++i) {
#if defined(QMM_HAVE_ASYNC_LDS)
            __builtin_amdgcn_global_load_async_to_lds_b128(
                (GV4*)(gA[i] + kb), (LV4*)(ldsA[buf] + soff[i]), 0, 0);
            __builtin_amdgcn_global_load_async_to_lds_b128(
                (GV4*)(gB[i] + kb), (LV4*)(ldsB[buf] + soff[i]), 0, 0);
#else
            *(uint4*)(ldsA[buf] + soff[i]) = *(const uint4*)(gA[i] + kb);
            *(uint4*)(ldsB[buf] + soff[i]) = *(const uint4*)(gB[i] + kb);
#endif
        }
    };

    auto compute = [&](int buf) {
        union { v16i v; unsigned long long q[8]; } afrag[4];
        #pragma unroll
        for (int mt = 0; mt < 4; ++mt) {
            const unsigned char* ap =
                ldsA[buf] + ((wm + mt) * 16 + row16) * 128 + hl * 16;
            #pragma unroll
            for (int c = 0; c < 8; ++c)
                afrag[mt].q[c] = *(const unsigned long long*)(ap + (c >> 1) * 32 + (c & 1) * 8);
        }
        #pragma unroll
        for (int nt = 0; nt < 2; ++nt) {
            union { v16i v; uint4 x[4]; } bfrag;
            const unsigned char* bp =
                ldsB[buf] + ((wn + nt) * 16 + row16) * 128 + hl * 16;
            #pragma unroll
            for (int g = 0; g < 4; ++g)
                bfrag.x[g] = *(const uint4*)(bp + g * 32);

            #pragma unroll
            for (int mt = 0; mt < 4; ++mt)
                acc[mt][nt] = __builtin_amdgcn_wmma_f32_16x16x128_fp8_fp8(
                    afrag[mt].v, bfrag.v, (short)0, acc[mt][nt],
                     false,  false);
        }
    };

#if defined(QMM_HAVE_ASYNC_LDS)
    stage(0, 0);
    for (int i = 0; i < nChunks; ++i) {
        const int cur = i & 1;
        if (i + 1 < nChunks) {
            stage((i + 1) << 7, cur ^ 1);
            qmm_wait_async_8();
        } else {
            qmm_wait_async_0();
        }
        __syncthreads();
        compute(cur);
        __syncthreads();
    }
#else
    for (int i = 0; i < nChunks; ++i) {
        const int cur = i & 1;
        stage(i << 7, cur);
        __syncthreads();
        compute(cur);
        __syncthreads();
    }
#endif

    const float sc = slot_scale(amax + 0) * slot_scale(amax + 1);

    #pragma unroll
    for (int mt = 0; mt < 4; ++mt)
        #pragma unroll
        for (int nt = 0; nt < 2; ++nt) {
            int col = nBase + (wn + nt) * 16 + row16;
            #pragma unroll
            for (int r = 0; r < 8; ++r) {
                int rowm = mBase + (wm + mt) * 16 + hl * 8 + r;
                out[batchC + (size_t)rowm * N + col] = acc[mt][nt][r] * sc;
            }
        }
}

extern "C" void kernel_launch(void* const* d_in, const int* in_sizes, int n_in,
                              void* d_out, int out_size, void* d_ws, size_t ws_size,
                              hipStream_t stream) {
    const float* in1 = (const float*)d_in[0];
    const float* in2 = (const float*)d_in[1];
    float* out = (float*)d_out;

    const int M = 4096, N = 4096, K = 4096;
    const long n1 = (long)in_sizes[0];
    const long n2 = (long)in_sizes[1];
    const int  B  = (int)(n1 / ((long)M * K));

    unsigned*      amax = (unsigned*)d_ws;
    unsigned char* q1   = (unsigned char*)d_ws + 256;
    unsigned char* q2t  = q1 + n1;

    qmm_init<<<1, 32, 0, stream>>>(amax);
    qmm_amax<<<2048, 256, 0, stream>>>((const float4*)in1, n1 / 4, amax + 0);
    qmm_amax<<<1024, 256, 0, stream>>>((const float4*)in2, n2 / 4, amax + 1);
    qmm_quant<<<(unsigned)((n1 / 16 + 255) / 256), 256, 0, stream>>>(
        (const float4*)in1, q1, amax + 0, n1 / 16);
    qmm_quant_t<<<dim3(N / 32, K / 32), dim3(32, 8), 0, stream>>>(
        in2, q2t, amax + 1, K, N);
    qmm_gemm<<<dim3(N / 128, M / 128, B), 256, 0, stream>>>(
        q1, q2t, out, amax, M, N, K);
}
